// GCMCLayer_23227183136844
// MI455X (gfx1250) — compile-verified, matrix operand order checked by reading, not run on hardware
//
#include <hip/hip_runtime.h>
#include <hip/hip_bf16.h>

#define R     5
#define NU    100000
#define NV    50000
#define NE    200000
#define DIN   100000
#define DSUB  32
#define DMSG  96
#define DOUT  64
#define HH    128
#define DREV  64

typedef __attribute__((ext_vector_type(16))) _Float16 v16h;
typedef __attribute__((ext_vector_type(8)))  _Float16 v8h;
typedef __attribute__((ext_vector_type(8)))  float    v8f;

#define WMMA_F16(a, b, c) \
  __builtin_amdgcn_wmma_f32_16x16x32_f16(false, (a), false, (b), (short)0, (c), false, false)

// ---- WMMA fragment loader (CDNA5 ISA 7.12.2 16-bit layout, wave32) ----
// Required element order: e=0..7 -> k = hf*8 + e ; e=8..15 -> k = 16 + hf*8 + (e-8),
// i.e. two contiguous 8-half (16 B) runs per lane -> two ds_load_b128.
// Works for A from row-major [m][k] tiles and for B from transpose-staged [n][k] tiles.
__device__ inline v16h frag_rowmajor(const _Float16* src, int ld) {
  int lane = threadIdx.x & 31, m = lane & 15, hf = lane >> 4;
  const _Float16* p = src + m * ld + hf * 8;
  union { v16h v; v8h h[2]; } u;
  u.h[0] = *(const v8h*)(p);
  u.h[1] = *(const v8h*)(p + 16);
  return u.v;
}

__device__ inline float lrelu(float x) { return x > 0.f ? x : 0.01f * x; }

// K1: fu[r,n,kk*32+d] = cj[n] * sum_b att[r,b]*basis[b, idx[n,kk], d]
__global__ void k_build_feat(const int* __restrict__ idx3, const float* __restrict__ cj,
                             const float* __restrict__ att, const float* __restrict__ basis,
                             float* __restrict__ out, int N) {
  size_t t = (size_t)blockIdx.x * blockDim.x + threadIdx.x;
  size_t total = (size_t)R * N * DMSG;
  if (t >= total) return;
  int    d  = (int)(t % DSUB);
  int    kk = (int)((t / DSUB) % 3);
  size_t n  = (t / DMSG) % N;
  int    r  = (int)(t / ((size_t)N * DMSG));
  int gi = idx3[n * 3 + kk];
  float s = 0.f;
#pragma unroll
  for (int b = 0; b < 5; ++b)
    s += att[r * 5 + b] * basis[((size_t)b * DIN + gi) * DSUB + d];
  out[t] = s * cj[n];
}

// K2: pa[r,e] = sigmoid(dot(review_feat[r,e,:], pw[r,:])), one wave per edge
__global__ __launch_bounds__(256) void k_edge_prob(const float* __restrict__ rf,
                                                   const float* __restrict__ pw,
                                                   float* __restrict__ pa) {
  int lane = threadIdx.x & 31;
  size_t eg = (size_t)blockIdx.x * 8 + (threadIdx.x >> 5);
  if (eg >= (size_t)R * NE) return;
  int r = (int)(eg / NE);
  const float* row = rf + eg * DREV;
  float s = row[lane] * pw[r * DREV + lane] + row[lane + 32] * pw[r * DREV + lane + 32];
#pragma unroll
  for (int o = 16; o; o >>= 1) s += __shfl_xor(s, o, 32);
  if (lane == 0) pa[eg] = 1.f / (1.f + __expf(-s));
}

// K3: accum[r, to[e], :] += pa[r,e] * feat[r, from[e], :] (one wave per edge)
__global__ __launch_bounds__(256) void k_scatter(const float* __restrict__ feat,
                                                 const float* __restrict__ pa,
                                                 const int* __restrict__ efrom,
                                                 const int* __restrict__ eto,
                                                 float* __restrict__ accum,
                                                 int Nfrom, int Nto) {
  int lane = threadIdx.x & 31;
  size_t eg = (size_t)blockIdx.x * 8 + (threadIdx.x >> 5);
  if (eg >= (size_t)R * NE) return;
  size_t r = eg / NE;
  int nf = efrom[eg], nt = eto[eg];
  float p = pa[eg];
  const float* f = feat + ((size_t)r * Nfrom + nf) * DMSG;
  float* a = accum + ((size_t)r * Nto + nt) * DMSG;
#pragma unroll
  for (int j = 0; j < 3; ++j) {
    int d = lane + 32 * j;
    atomicAdd(&a[d], p * f[d]);
  }
}

// K3b: buf[r,n,d] *= ci[n]
__global__ void k_scale(float* __restrict__ buf, const float* __restrict__ ci, int N) {
  size_t t = (size_t)blockIdx.x * blockDim.x + threadIdx.x;
  size_t total = (size_t)R * N * DMSG;
  if (t >= total) return;
  size_t n = (t / DMSG) % N;
  buf[t] *= ci[n];
}

// K4/K9: crossrate over 5 ratings + sum over r + FC. One 96-thread block per node.
__global__ __launch_bounds__(96) void k_crossrate(const float* __restrict__ z,
                                                  const float* __restrict__ w1,
                                                  const float* __restrict__ w2,
                                                  const float* __restrict__ fcw,
                                                  const float* __restrict__ fcb,
                                                  float* __restrict__ out, int N,
                                                  int applyLeaky) {
  const int nbr1[R] = {1, 2, 3, 4, 3};
  const int nbr2[R] = {2, 0, 1, 2, 2};
  int n = blockIdx.x, tid = threadIdx.x;
  __shared__ float sz[R * DMSG];
  __shared__ float sw1[DMSG * 16];
  __shared__ float sw2[16];
  __shared__ float ssc[R * 3];
  __shared__ float sbeta[R * 2];
  __shared__ float sout[DMSG];
  for (int r = 0; r < R; ++r) sz[r * DMSG + tid] = z[((size_t)r * N + n) * DMSG + tid];
  for (int t = tid; t < DMSG * 16; t += 96) sw1[t] = w1[t];
  if (tid < 16) sw2[tid] = w2[tid];
  if (tid < R * 3) ssc[tid] = 0.f;
  __syncthreads();
  // attention scores: 15 (r,nbr) pairs x 16 hidden units
  for (int t = tid; t < R * 3 * 16; t += 96) {
    int p = t >> 4, k = t & 15;
    int r = p / 3, j = p % 3;
    int zr = (j == 0) ? r : (j == 1 ? nbr1[r] : nbr2[r]);
    float h = 0.f;
    const float* zz = &sz[zr * DMSG];
    for (int d = 0; d < DMSG; ++d) h += zz[d] * sw1[d * 16 + k];
    h = lrelu(h);
    atomicAdd(&ssc[p], h * sw2[k]);
  }
  __syncthreads();
  if (tid < R) {
    float a0 = ssc[tid * 3], a1 = ssc[tid * 3 + 1], a2 = ssc[tid * 3 + 2];
    float m = fmaxf(a0, fmaxf(a1, a2));
    float e0 = __expf(a0 - m), e1 = __expf(a1 - m), e2 = __expf(a2 - m);
    float s = e0 + e1 + e2;
    sbeta[tid * 2] = e1 / s;
    sbeta[tid * 2 + 1] = e2 / s;
  }
  __syncthreads();
  {
    float o = 0.f;
    for (int r = 0; r < R; ++r)
      o += sz[r * DMSG + tid] + sbeta[r * 2] * sz[nbr1[r] * DMSG + tid] +
           sbeta[r * 2 + 1] * sz[nbr2[r] * DMSG + tid];
    sout[tid] = o;
  }
  __syncthreads();
  if (tid < DOUT) {
    float a = fcb[tid];
    for (int d = 0; d < DMSG; ++d) a += sout[d] * fcw[d * DOUT + tid];
    if (applyLeaky) a = lrelu(a);
    out[(size_t)n * DOUT + tid] = a;
  }
}

// K5: lat[r,h,d] += sum_n lrelu(h[r]@Hy[r])[n,h] * h[r,n,d]
// Two chained WMMA GEMMs; lat partials live in registers across the N loop,
// flushed once via global atomics. All B operands transpose-staged -> b128 LDS loads.
__global__ __launch_bounds__(256) void k_lat_accum(const float* __restrict__ hbuf,
                                                   const float* __restrict__ hyper,
                                                   float* __restrict__ lat, int N) {
  const int r = blockIdx.y;
  __shared__ __align__(16) _Float16 sWt[HH * DMSG];  // hyper^T: [h][d]  (24 KB)
  __shared__ __align__(16) _Float16 sA [32 * DMSG];  // h tile  [n][d]  (6 KB)
  __shared__ __align__(16) _Float16 sAt[DMSG * 32];  // h tile^T [d][n] (6 KB)
  __shared__ __align__(16) _Float16 sUt[HH * 32];    // uu^T    [h][n]  (8 KB)
  const int tid = threadIdx.x, lane = tid & 31, w = tid >> 5;
  const float* Wg = hyper + (size_t)r * DMSG * HH;   // [d][h]
  for (int t = tid; t < DMSG * HH; t += 256) {       // coalesced global, strided LDS
    int h = t % HH, d = t / HH;
    sWt[h * DMSG + d] = (_Float16)Wg[t];
  }

  v8f acc2[6];
#pragma unroll
  for (int j = 0; j < 6; ++j)
#pragma unroll
    for (int e = 0; e < 8; ++e) acc2[j][e] = 0.f;

  const int step = gridDim.x * 32;
  for (int n0 = blockIdx.x * 32; n0 < N; n0 += step) {
    __syncthreads();  // previous GEMM2 reads done
    const float* Hg = hbuf + ((size_t)r * N + n0) * DMSG;
    for (int t = tid; t < 32 * DMSG; t += 256) {
      int row = t / DMSG, d = t % DMSG;
      _Float16 v = (n0 + row < N) ? (_Float16)Hg[t] : (_Float16)0.f;
      sA[t] = v;
      sAt[d * 32 + row] = v;
    }
    // prefetch next tile (global_prefetch_b8), one 128B line per thread
    if (n0 + step < N && tid < 96)
      __builtin_prefetch((const char*)(hbuf + ((size_t)r * N + n0 + step) * DMSG) + tid * 128, 0, 3);
    __syncthreads();
    // GEMM1: U(32x128) = lrelu(A(32x96) @ W(96x128)); wave w owns h-block nt=w
    for (int id = w; id < 16; id += 8) {
      int mt = id >> 3, nt = id & 7;
      v8f c = {};
#pragma unroll
      for (int kt = 0; kt < 3; ++kt) {
        v16h a = frag_rowmajor(sA + mt * 16 * DMSG + kt * 32, DMSG);
        v16h b = frag_rowmajor(sWt + (nt * 16) * DMSG + kt * 32, DMSG);
        c = WMMA_F16(a, b, c);
      }
      int nc = lane & 15, hf = lane >> 4;
#pragma unroll
      for (int e = 0; e < 8; ++e)
        sUt[(nt * 16 + nc) * 32 + mt * 16 + e + hf * 8] = (_Float16)lrelu(c[e]);
    }
    __syncthreads();
    // GEMM2: lat_part(128x96) += U^T(128x32) @ A(32x96); wave w owns h-block w
    v16h a2 = frag_rowmajor(sUt + (w * 16) * 32, 32);
#pragma unroll
    for (int j = 0; j < 6; ++j) {
      v16h b = frag_rowmajor(sAt + (j * 16) * 32, 32);
      acc2[j] = WMMA_F16(a2, b, acc2[j]);
    }
  }
  __syncthreads();
  {
    int nc = lane & 15, hf = lane >> 4;
    float* L = lat + (size_t)r * HH * DMSG;
    for (int j = 0; j < 6; ++j)
#pragma unroll
      for (int e = 0; e < 8; ++e)
        atomicAdd(&L[(w * 16 + e + hf * 8) * DMSG + j * 16 + nc], acc2[j][e]);
  }
}

// K6: softmax over the rating axis of lat (R,128,96), in place
__global__ void k_softmax_r(float* __restrict__ lat) {
  int t = blockIdx.x * blockDim.x + threadIdx.x;
  if (t >= HH * DMSG) return;
  float v[R], m = -3.4e38f;
#pragma unroll
  for (int r = 0; r < R; ++r) { v[r] = lat[r * HH * DMSG + t]; m = fmaxf(m, v[r]); }
  float s = 0.f;
#pragma unroll
  for (int r = 0; r < R; ++r) { v[r] = __expf(v[r] - m); s += v[r]; }
#pragma unroll
  for (int r = 0; r < R; ++r) lat[r * HH * DMSG + t] = v[r] / s;
}

// K7: g[r,n,d] = sum_h lrelu(h[r]@Hy[r])[n,h] * lat[r,h,d]  (uu recomputed, two WMMA GEMMs)
__global__ __launch_bounds__(256) void k_hgnn_out(const float* __restrict__ hbuf,
                                                  const float* __restrict__ hyper,
                                                  const float* __restrict__ lat,
                                                  float* __restrict__ g, int N) {
  const int r = blockIdx.y, n0 = blockIdx.x * 32;
  __shared__ __align__(16) _Float16 sWt[HH * DMSG];  // hyper^T [h][d]  24 KB
  __shared__ __align__(16) _Float16 sLt[DMSG * HH];  // lat^T   [d][h]  24 KB
  __shared__ __align__(16) _Float16 sA [32 * DMSG];  // h tile  [n][d]   6 KB
  __shared__ __align__(16) _Float16 sU [32 * HH];    // uu      [n][h]   8 KB
  const int tid = threadIdx.x, lane = tid & 31, w = tid >> 5;
  const float* Wg = hyper + (size_t)r * DMSG * HH;   // [d][h]
  const float* Lg = lat + (size_t)r * HH * DMSG;     // [h][d]
  for (int t = tid; t < DMSG * HH; t += 256) {
    int h = t % HH, d = t / HH;
    sWt[h * DMSG + d] = (_Float16)Wg[t];             // t == d*HH + h
  }
  for (int t = tid; t < HH * DMSG; t += 256) {
    int h = t / DMSG, d = t % DMSG;
    sLt[d * HH + h] = (_Float16)Lg[t];
  }
  const float* Hg = hbuf + ((size_t)r * N + n0) * DMSG;
  for (int t = tid; t < 32 * DMSG; t += 256) {
    int row = t / DMSG;
    sA[t] = (n0 + row < N) ? (_Float16)Hg[t] : (_Float16)0.f;
  }
  __syncthreads();
  for (int id = w; id < 16; id += 8) {  // GEMM1: U = lrelu(A @ W)
    int mt = id >> 3, nt = id & 7;
    v8f c = {};
#pragma unroll
    for (int kt = 0; kt < 3; ++kt) {
      v16h a = frag_rowmajor(sA + mt * 16 * DMSG + kt * 32, DMSG);
      v16h b = frag_rowmajor(sWt + (nt * 16) * DMSG + kt * 32, DMSG);
      c = WMMA_F16(a, b, c);
    }
    int nc = lane & 15, hf = lane >> 4;
#pragma unroll
    for (int e = 0; e < 8; ++e)
      sU[(mt * 16 + e + hf * 8) * HH + nt * 16 + nc] = (_Float16)lrelu(c[e]);
  }
  __syncthreads();
  for (int id = w; id < 12; id += 8) {  // GEMM2: out(32x96) = U(32x128) @ L(128x96)
    int mt = id / 6, nt = id % 6;
    v8f c = {};
#pragma unroll
    for (int kt = 0; kt < 4; ++kt) {
      v16h a = frag_rowmajor(sU + mt * 16 * HH + kt * 32, HH);
      v16h b = frag_rowmajor(sLt + (nt * 16) * HH + kt * 32, HH);
      c = WMMA_F16(a, b, c);
    }
    int nc = lane & 15, hf = lane >> 4;
#pragma unroll
    for (int e = 0; e < 8; ++e) {
      int row = n0 + mt * 16 + e + hf * 8;
      if (row < N) g[((size_t)r * N + row) * DMSG + nt * 16 + nc] = c[e];
    }
  }
}

// K8a: per-(r,d) online (max,sumexp) over the N axis
__global__ __launch_bounds__(256) void k_colstats(const float* __restrict__ g,
                                                  float* __restrict__ cmax,
                                                  float* __restrict__ csum, int N) {
  int r = blockIdx.x / DMSG, d = blockIdx.x % DMSG, tid = threadIdx.x;
  float m = -3.4e38f, s = 0.f;
  for (int n = tid; n < N; n += 256) {
    float x = g[((size_t)r * N + n) * DMSG + d];
    if (x > m) { s = s * __expf(m - x) + 1.f; m = x; } else s += __expf(x - m);
  }
  __shared__ float sm[256], ss[256];
  sm[tid] = m; ss[tid] = s;
  __syncthreads();
  for (int o = 128; o; o >>= 1) {
    if (tid < o) {
      float m2 = sm[tid + o], s2 = ss[tid + o];
      float mm = fmaxf(sm[tid], m2);
      ss[tid] = ss[tid] * __expf(sm[tid] - mm) + s2 * __expf(m2 - mm);
      sm[tid] = mm;
    }
    __syncthreads();
  }
  if (tid == 0) { cmax[blockIdx.x] = sm[0]; csum[blockIdx.x] = ss[0]; }
}

// K8b: normalize (softmax over N) in place
__global__ void k_colnorm(float* __restrict__ g, const float* __restrict__ cmax,
                          const float* __restrict__ csum, int N) {
  size_t t = (size_t)blockIdx.x * blockDim.x + threadIdx.x;
  size_t total = (size_t)R * N * DMSG;
  if (t >= total) return;
  int d = (int)(t % DMSG);
  int r = (int)(t / ((size_t)N * DMSG));
  g[t] = __expf(g[t] - cmax[r * DMSG + d]) / csum[r * DMSG + d];
}

extern "C" void kernel_launch(void* const* d_in, const int* in_sizes, int n_in,
                              void* d_out, int out_size, void* d_ws, size_t ws_size,
                              hipStream_t stream) {
  (void)in_sizes; (void)n_in; (void)out_size; (void)ws_size;
  const int*   ufeat_idx = (const int*)d_in[0];
  const int*   ifeat_idx = (const int*)d_in[1];
  const int*   edge_src  = (const int*)d_in[2];
  const int*   edge_dst  = (const int*)d_in[3];
  const float* review    = (const float*)d_in[4];
  const float* cj_u      = (const float*)d_in[5];
  const float* ci_u      = (const float*)d_in[6];
  const float* cj_v      = (const float*)d_in[7];
  const float* ci_v      = (const float*)d_in[8];
  const float* att       = (const float*)d_in[9];
  const float* basis     = (const float*)d_in[10];
  const float* pwf       = (const float*)d_in[11];
  const float* pwr       = (const float*)d_in[12];
  const float* w1        = (const float*)d_in[13];
  const float* w2        = (const float*)d_in[14];
  const float* ufc_w     = (const float*)d_in[15];
  const float* ufc_b     = (const float*)d_in[16];
  const float* ufc2_w    = (const float*)d_in[17];
  const float* ufc2_b    = (const float*)d_in[18];
  const float* uHyper    = (const float*)d_in[19];
  const float* iHyper    = (const float*)d_in[20];
  float* out = (float*)d_out;

  // workspace carve (floats)
  float* ws = (float*)d_ws;
  size_t off = 0;
  float* fu    = ws + off; off += (size_t)R * NU * DMSG;   // reused as g_u
  float* fv    = ws + off; off += (size_t)R * NV * DMSG;   // reused as g_v
  float* hu    = ws + off; off += (size_t)R * NU * DMSG;
  float* hv    = ws + off; off += (size_t)R * NV * DMSG;
  float* pa_f  = ws + off; off += (size_t)R * NE;
  float* pa_r  = ws + off; off += (size_t)R * NE;
  float* lat_u = ws + off; off += (size_t)R * HH * DMSG;
  float* lat_v = ws + off; off += (size_t)R * HH * DMSG;
  float* cmax_u = ws + off; off += (size_t)R * DMSG;
  float* csum_u = ws + off; off += (size_t)R * DMSG;
  float* cmax_v = ws + off; off += (size_t)R * DMSG;
  float* csum_v = ws + off; off += (size_t)R * DMSG;
  float* g_u = fu;  // fu/fv dead after the scatter phase
  float* g_v = fv;

  hipMemsetAsync(hu, 0, (size_t)R * NU * DMSG * sizeof(float), stream);
  hipMemsetAsync(hv, 0, (size_t)R * NV * DMSG * sizeof(float), stream);
  hipMemsetAsync(lat_u, 0, (size_t)R * HH * DMSG * sizeof(float), stream);
  hipMemsetAsync(lat_v, 0, (size_t)R * HH * DMSG * sizeof(float), stream);

  // K1: gathered basis blend (basis is L2-resident, 64 MB)
  {
    size_t tot = (size_t)R * NU * DMSG;
    k_build_feat<<<(unsigned)((tot + 255) / 256), 256, 0, stream>>>(ufeat_idx, cj_u, att, basis, fu, NU);
    tot = (size_t)R * NV * DMSG;
    k_build_feat<<<(unsigned)((tot + 255) / 256), 256, 0, stream>>>(ifeat_idx, cj_v, att, basis, fv, NV);
  }
  // K2: edge gates
  k_edge_prob<<<(R * NE) / 8, 256, 0, stream>>>(review, pwf, pa_f);
  k_edge_prob<<<(R * NE) / 8, 256, 0, stream>>>(review, pwr, pa_r);
  // K3: segment sums (atomic scatter), then ci scaling
  k_scatter<<<(R * NE) / 8, 256, 0, stream>>>(fu, pa_f, edge_src, edge_dst, hv, NU, NV);
  k_scatter<<<(R * NE) / 8, 256, 0, stream>>>(fv, pa_r, edge_dst, edge_src, hu, NV, NU);
  {
    size_t tot = (size_t)R * NU * DMSG;
    k_scale<<<(unsigned)((tot + 255) / 256), 256, 0, stream>>>(hu, ci_u, NU);
    tot = (size_t)R * NV * DMSG;
    k_scale<<<(unsigned)((tot + 255) / 256), 256, 0, stream>>>(hv, ci_v, NV);
  }
  // K4: crossrate + FC -> u_out, v_out
  k_crossrate<<<NU, 96, 0, stream>>>(hu, w1, w2, ufc_w, ufc_b, out, NU, 1);
  k_crossrate<<<NV, 96, 0, stream>>>(hv, w1, w2, ufc_w, ufc_b, out + (size_t)NU * DOUT, NV, 1);
  // K5: lat = (lrelu(h@Hy))^T @ h   (WMMA, register-accumulated over N)
  k_lat_accum<<<dim3(128, R), 256, 0, stream>>>(hu, uHyper, lat_u, NU);
  k_lat_accum<<<dim3(128, R), 256, 0, stream>>>(hv, iHyper, lat_v, NV);
  // K6: softmax over r
  k_softmax_r<<<(HH * DMSG + 255) / 256, 256, 0, stream>>>(lat_u);
  k_softmax_r<<<(HH * DMSG + 255) / 256, 256, 0, stream>>>(lat_v);
  // K7: g = lrelu(h@Hy) @ lat   (WMMA, uu recomputed on the fly)
  k_hgnn_out<<<dim3((NU + 31) / 32, R), 256, 0, stream>>>(hu, uHyper, lat_u, g_u, NU);
  k_hgnn_out<<<dim3((NV + 31) / 32, R), 256, 0, stream>>>(hv, iHyper, lat_v, g_v, NV);
  // K8: softmax over the N axis
  k_colstats<<<R * DMSG, 256, 0, stream>>>(g_u, cmax_u, csum_u, NU);
  k_colstats<<<R * DMSG, 256, 0, stream>>>(g_v, cmax_v, csum_v, NV);
  {
    size_t tot = (size_t)R * NU * DMSG;
    k_colnorm<<<(unsigned)((tot + 255) / 256), 256, 0, stream>>>(g_u, cmax_u, csum_u, NU);
    tot = (size_t)R * NV * DMSG;
    k_colnorm<<<(unsigned)((tot + 255) / 256), 256, 0, stream>>>(g_v, cmax_v, csum_v, NV);
  }
  // K9: crossrate + FC2 (no activation) -> uh_out, ih_out
  k_crossrate<<<NU, 96, 0, stream>>>(g_u, w1, w2, ufc2_w, ufc2_b,
                                     out + (size_t)(NU + NV) * DOUT, NU, 0);
  k_crossrate<<<NV, 96, 0, stream>>>(g_v, w1, w2, ufc2_w, ufc2_b,
                                     out + (size_t)(2 * NU + NV) * DOUT, NV, 0);
}
